// ResidualQuantizer_80728205296119
// MI455X (gfx1250) — matrix operand order repaired and checked by reading, not compiled, not run on hardware
//
#include <hip/hip_runtime.h>

// ---------------------------------------------------------------------------
// Residual quantizer encode for MI455X (gfx1250, wave32, WMMA).
//   x:         (262144, 64) f32
//   codebooks: (8, 256, 64) f32
//   out codes: (262144, 8)  int32
// Scoring GEMM on v_wmma_f32_16x16x32_f16 (f16 in, f32 accumulate) with the
// -0.5*||c||^2 bias folded into the WMMA C operand, so the accumulator is
// r.c - 0.5||c||^2 and argmax(acc) == argmin(distance). Residual kept in
// fp32 in LDS across the 8 sequential levels. Per-level codebook fragments
// and norm biases staged to LDS via global_load_async_to_lds_b128.
// ---------------------------------------------------------------------------

typedef __attribute__((ext_vector_type(16))) _Float16 v16h;
typedef __attribute__((ext_vector_type(8)))  float    v8f;

#define N_ROWS   262144
#define DIM      64
#define NLEV     8
#define KCODES   256
#define WAVES    8          // waves per block
#define ROWS_PB  (WAVES*16) // 128 rows per block

// Packed B-fragment layout (contiguous k order, 16-bit B 32x16):
//   pk[ ((level*16 + t)*2 + c)*512 + lane*16 + j ]
//     = f16( cb[level][ t*16 + (lane&15) ][ c*32 + (lane>>4)*16 + j ] )
// Each lane's 16 f16 (32B) are contiguous.
#define PK_LEVEL   16384                    // f16 elems per level (32 KB)
#define PK_ELEMS   (NLEV*PK_LEVEL)          // 131072 f16 = 256 KB
#define PK_BYTES   (PK_ELEMS*2)

__global__ __launch_bounds__(256)
void rq_pack_kernel(const float* __restrict__ cb,
                    _Float16* __restrict__ pk,
                    float* __restrict__ nbias) {
    int p = blockIdx.x * 256 + threadIdx.x;          // [0, 131072)
    int j     =  p        & 15;
    int lane  = (p >> 4)  & 31;
    int c     = (p >> 9)  & 1;
    int t     = (p >> 10) & 15;
    int level =  p >> 14;
    int n = t * 16 + (lane & 15);
    int K = c * 32 + (lane >> 4) * 16 + j;
    pk[p] = (_Float16)cb[(level * KCODES + n) * DIM + K];

    if (p < NLEV * KCODES) {                         // -0.5*||c_k||^2
        const float* row = cb + p * DIM;
        float s = 0.f;
        #pragma unroll 8
        for (int i = 0; i < DIM; ++i) s += row[i] * row[i];
        nbias[p] = -0.5f * s;
    }
}

__global__ __launch_bounds__(256)
void rq_encode_kernel(const float* __restrict__ x,
                      const float* __restrict__ cb,
                      const _Float16* __restrict__ pk,
                      const float* __restrict__ nbias,
                      int* __restrict__ out) {
    __shared__ __align__(16) float    resid[WAVES * 16 * DIM];  // 32 KB
    __shared__ __align__(16) _Float16 bstage[PK_LEVEL];         // 32 KB
    __shared__ __align__(16) float    nstage[KCODES];           // 1 KB
    __shared__ int idxLds[WAVES * 16];

    const int tid   = threadIdx.x;
    const int wave  = tid >> 5;
    const int lane  = tid & 31;
    const int gbase = blockIdx.x * ROWS_PB;

    // ---- stage x rows into LDS residual (fp32) --------------------------
    {
        const int rr = lane >> 1, ch = lane & 1;     // 2 lanes per row
        const float4* src = (const float4*)(x + (size_t)(gbase + wave * 16 + rr) * DIM + ch * 32);
        float4* dst = (float4*)(resid + wave * 1024 + rr * DIM + ch * 32);
        #pragma unroll
        for (int i = 0; i < 8; ++i) dst[i] = src[i];
    }

    const int m = lane & 15;        // row (A) / column (B,C) within tile
    const int h = lane >> 4;        // lane-half
    const float* rrow = resid + wave * 1024 + m * DIM;
    const unsigned ldsb = (unsigned)(uintptr_t)bstage;
    const unsigned ldsn = (unsigned)(uintptr_t)nstage;

    for (int level = 0; level < NLEV; ++level) {
        // ---- async-stage this level's packed B (32 KB) + biases (1 KB) --
        {
            const char* gb = (const char*)(pk + (size_t)level * PK_LEVEL);
            #pragma unroll
            for (int i = 0; i < 8; ++i) {
                unsigned laddr = ldsb + (unsigned)(tid * 16 + i * 4096);
                unsigned long long ga =
                    (unsigned long long)(uintptr_t)(gb + tid * 16 + i * 4096);
                asm volatile("global_load_async_to_lds_b128 %0, %1, off"
                             :: "v"(laddr), "v"(ga) : "memory");
            }
            if (wave < 2) {         // waves 0-1: 64 lanes x 16B = 1 KB biases
                unsigned laddr = ldsn + (unsigned)(tid * 16);
                unsigned long long ga =
                    (unsigned long long)(uintptr_t)(nbias + level * KCODES + tid * 4);
                asm volatile("global_load_async_to_lds_b128 %0, %1, off"
                             :: "v"(laddr), "v"(ga) : "memory");
            }
            asm volatile("s_wait_asynccnt 0x0" ::: "memory");
        }
        if (level + 1 < NLEV)       // warm L2 for next level's staging
            __builtin_prefetch(pk + (size_t)(level + 1) * PK_LEVEL + tid * 8, 0, 2);
        __syncthreads();            // residual (lvl 0: x) + stages visible

        // ---- build A fragments (16x32 f16, contiguous k order) -------------
        // chunk c: lane holds K = c*32 + h*16 + [0..16)
        // Load all 8 float4 first (one ds clause, one wait), then convert.
        float4 f[8];
        #pragma unroll
        for (int c = 0; c < 2; ++c) {
            const float4* p0 = (const float4*)(rrow + c * 32 + h * 16);
            const float4* p1 = (const float4*)(rrow + c * 32 + h * 16 + 8);
            f[c * 4 + 0] = p0[0];
            f[c * 4 + 1] = p0[1];
            f[c * 4 + 2] = p1[0];
            f[c * 4 + 3] = p1[1];
        }
        v16h afrag[2];
        #pragma unroll
        for (int c = 0; c < 2; ++c) {
            const float4 g0 = f[c*4+0], g1 = f[c*4+1], g2 = f[c*4+2], g3 = f[c*4+3];
            v16h a;
            a[0]=(_Float16)g0.x; a[1]=(_Float16)g0.y; a[2]=(_Float16)g0.z; a[3]=(_Float16)g0.w;
            a[4]=(_Float16)g1.x; a[5]=(_Float16)g1.y; a[6]=(_Float16)g1.z; a[7]=(_Float16)g1.w;
            a[8]=(_Float16)g2.x; a[9]=(_Float16)g2.y; a[10]=(_Float16)g2.z; a[11]=(_Float16)g2.w;
            a[12]=(_Float16)g3.x; a[13]=(_Float16)g3.y; a[14]=(_Float16)g3.z; a[15]=(_Float16)g3.w;
            afrag[c] = a;
        }

        // ---- preload per-lane -0.5*||c||^2 for all 16 tiles (from LDS) --
        float nrm[16];
        #pragma unroll
        for (int t = 0; t < 16; ++t) nrm[t] = nstage[t * 16 + m];

        float bestv[8];
        int   besti[8];
        #pragma unroll
        for (int r = 0; r < 8; ++r) { bestv[r] = -3.402823e38f; besti[r] = 0; }

        // ---- 16 column tiles of 16 codewords, B from LDS, fully unrolled
        #pragma unroll
        for (int t = 0; t < 16; ++t) {
            v16h b0 = *(const v16h*)(bstage + t * 1024 +       lane * 16);
            v16h b1 = *(const v16h*)(bstage + t * 1024 + 512 + lane * 16);

            // C init = -0.5*||c_k||^2 (same value for all 8 rows this lane)
            v8f acc;
            #pragma unroll
            for (int r = 0; r < 8; ++r) acc[r] = nrm[t];

            acc = __builtin_amdgcn_wmma_f32_16x16x32_f16(
                    false, afrag[0], false, b0, (short)0, acc, false, false);
            acc = __builtin_amdgcn_wmma_f32_16x16x32_f16(
                    false, afrag[1], false, b1, (short)0, acc, false, false);

            // argmax(r.c - 0.5||c||^2) == argmin(||r - c||^2); ascending t
            // with strict '>' keeps the earliest index on ties.
            const int kidx = t * 16 + m;
            #pragma unroll
            for (int r = 0; r < 8; ++r) {
                if (acc[r] > bestv[r]) { bestv[r] = acc[r]; besti[r] = kidx; }
            }
        }

        // ---- argmax across the 16 lanes of each half (wave32 shfl) ------
        #pragma unroll
        for (int mask = 1; mask < 16; mask <<= 1) {
            #pragma unroll
            for (int r = 0; r < 8; ++r) {
                float ov = __shfl_xor(bestv[r], mask, 32);
                int   oi = __shfl_xor(besti[r], mask, 32);
                if (ov > bestv[r] || (ov == bestv[r] && oi < besti[r])) {
                    bestv[r] = ov; besti[r] = oi;
                }
            }
        }

        if (m == 0) {   // lanes 0 and 16 own rows h*8 + [0..8)
            #pragma unroll
            for (int r = 0; r < 8; ++r) {
                int row = h * 8 + r;
                idxLds[wave * 16 + row] = besti[r];
                out[(size_t)(gbase + wave * 16 + row) * NLEV + level] = besti[r];
            }
        }
        __syncthreads();

        // ---- residual -= chosen codeword (fp32, skip after last level) --
        if (level < NLEV - 1) {
            const int rr = lane >> 1, ch = lane & 1;
            const int k  = idxLds[wave * 16 + rr];
            const float4* c4 = (const float4*)(cb + ((size_t)(level * KCODES + k)) * DIM + ch * 32);
            float4* r4 = (float4*)(resid + wave * 1024 + rr * DIM + ch * 32);
            #pragma unroll
            for (int i = 0; i < 8; ++i) {
                float4 rv = r4[i], cv = c4[i];
                rv.x -= cv.x; rv.y -= cv.y; rv.z -= cv.z; rv.w -= cv.w;
                r4[i] = rv;
            }
            __syncthreads();
        }
    }
}

extern "C" void kernel_launch(void* const* d_in, const int* in_sizes, int n_in,
                              void* d_out, int out_size, void* d_ws, size_t ws_size,
                              hipStream_t stream) {
    const float* x  = (const float*)d_in[0];
    const float* cb = (const float*)d_in[1];
    int* out        = (int*)d_out;

    _Float16* pk = (_Float16*)d_ws;                       // 256 KB packed f16 B-frags
    float* nbias = (float*)((char*)d_ws + PK_BYTES);      // 8 KB  -0.5*||c||^2

    // 1) pack codebooks into WMMA B-fragment layout + biased norms
    rq_pack_kernel<<<PK_ELEMS / 256, 256, 0, stream>>>(cb, pk, nbias);

    // 2) fused 8-level encode
    rq_encode_kernel<<<N_ROWS / ROWS_PB, 256, 0, stream>>>(x, cb, pk, nbias, out);
}
